// Attention_13125420057022
// MI455X (gfx1250) — matrix operand order repaired and checked by reading, not compiled, not run on hardware
//
#include <hip/hip_runtime.h>

// ---------------------------------------------------------------------------
// Attention layer (B=2, S=2048, H=768, 12 heads, D=64) for gfx1250 (CDNA5).
// All GEMMs use v_wmma_f32_16x16x32_bf16 (bf16 inputs, f32 accumulate).
// Dense GEMMs use 64x64 register tiles (16 accumulators) per wave.
// ---------------------------------------------------------------------------

#define BATCH 2
#define SEQ 2048
#define HIDDEN 768
#define NHEAD 12
#define HDIM 64

typedef __attribute__((ext_vector_type(16))) __bf16 v16bf;
typedef __attribute__((ext_vector_type(8)))  float  v8f;

struct U32B { uint4 lo, hi; };  // 32 bytes = 16 bf16

static __device__ __forceinline__ v16bf pack16(uint4 lo, uint4 hi) {
    U32B t{lo, hi};
    return __builtin_bit_cast(v16bf, t);
}

// float -> bf16 (round to nearest even-ish)
static __device__ __forceinline__ unsigned short f2bf(float f) {
    unsigned u = __float_as_uint(f);
    u += 0x7FFFu + ((u >> 16) & 1u);
    return (unsigned short)(u >> 16);
}
static __device__ __forceinline__ float bf2f(unsigned short s) {
    return __uint_as_float(((unsigned)s) << 16);
}

// A-operand (16x32, bf16): lane holds row M = lane&15.
//   halves 0..7  = K = k0 + (lane>>4)*8 + 0..7
//   halves 8..15 = K = k0 + 16 + (lane>>4)*8 + 0..7
static __device__ __forceinline__ v16bf loadA(const unsigned short* rowp, int k0, int hi) {
    const unsigned short* p = rowp + k0 + hi * 8;
    uint4 lo = *(const uint4*)(p);
    uint4 h2 = *(const uint4*)(p + 16);
    return pack16(lo, h2);
}

// B-operand (32x16, bf16): lane holds column N = lane&15.
//   halves 0..7  = K = k0 + (lane>>4)*8 + 0..7
//   halves 8..15 = K = k0 + 16 + (lane>>4)*8 + 0..7
static __device__ __forceinline__ v16bf loadB(const unsigned short* rowp, int k0, int hi) {
    const unsigned short* p = rowp + k0 + hi * 8;
    uint4 lo = *(const uint4*)(p);
    uint4 h2 = *(const uint4*)(p + 16);
    return pack16(lo, h2);
}

static __device__ __forceinline__ v8f wmma_bf16(v16bf a, v16bf b, v8f c) {
    return __builtin_amdgcn_wmma_f32_16x16x32_bf16(
        false, a, false, b, (short)0, c, false, false);
}

// ---------------------------------------------------------------------------
// fp32 -> bf16 conversion (grid-stride)
// ---------------------------------------------------------------------------
__global__ void cvt_bf16_kernel(const float* __restrict__ src,
                                unsigned short* __restrict__ dst, int n) {
    for (int i = blockIdx.x * blockDim.x + threadIdx.x; i < n;
         i += gridDim.x * blockDim.x)
        dst[i] = f2bf(src[i]);
}

// ---------------------------------------------------------------------------
// QKV GEMM: qkv[t,f] = sum_k x[t,k]*W[f,k] + b[f], 64x64 tile per wave,
// then RoPE on q,k and scatter to q[B,H,S,D], k[B,H,S,D], vT[B,H,D,S] (bf16).
// grid = (64 M-tiles of 64 tokens, 36 feature groups of 64), block = 32.
// ---------------------------------------------------------------------------
__global__ __launch_bounds__(32)
void qkv_rope_kernel(const unsigned short* __restrict__ xb,
                     const unsigned short* __restrict__ wb,
                     const float* __restrict__ bias,
                     unsigned short* __restrict__ q,
                     unsigned short* __restrict__ k,
                     unsigned short* __restrict__ vT) {
    const int mt = blockIdx.x;          // 64-token tile
    const int g  = blockIdx.y;          // 64-feature group: which = g/12, h = g%12
    const int lane = threadIdx.x;
    const int Mloc = lane & 15, hi = lane >> 4;
    const int t0 = mt * 64;

    const unsigned short* arow[4];
#pragma unroll
    for (int i = 0; i < 4; ++i)
        arow[i] = xb + (size_t)(t0 + i * 16 + Mloc) * HIDDEN;
    const unsigned short* wrow[4];
#pragma unroll
    for (int nt = 0; nt < 4; ++nt)
        wrow[nt] = wb + (size_t)(g * 64 + nt * 16 + Mloc) * HIDDEN;

    v8f acc[4][4];
#pragma unroll
    for (int i = 0; i < 4; ++i)
#pragma unroll
        for (int nt = 0; nt < 4; ++nt) acc[i][nt] = v8f{};

    for (int k0 = 0; k0 < HIDDEN; k0 += 32) {
        v16bf a[4];
#pragma unroll
        for (int i = 0; i < 4; ++i) {
            a[i] = loadA(arow[i], k0, hi);
            __builtin_prefetch(arow[i] + k0 + 128, 0, 1);   // global_prefetch_b8
        }
#pragma unroll
        for (int nt = 0; nt < 4; ++nt) {
            v16bf b = loadB(wrow[nt], k0, hi);
#pragma unroll
            for (int i = 0; i < 4; ++i)
                acc[i][nt] = wmma_bf16(a[i], b, acc[i][nt]);
        }
    }

    __shared__ float tile[64 * 64];   // 16 KB
#pragma unroll
    for (int i = 0; i < 4; ++i)
#pragma unroll
        for (int nt = 0; nt < 4; ++nt)
#pragma unroll
            for (int r = 0; r < 8; ++r)
                tile[(i * 16 + r + hi * 8) * 64 + nt * 16 + Mloc] = acc[i][nt][r];
    __syncthreads();

    const int which = g / NHEAD;        // 0=q 1=k 2=v
    const int h = g % NHEAD;

    for (int pp = lane; pp < 64 * 32; pp += 32) {   // 64 rows * 32 pairs
        const int m  = pp >> 5;
        const int pr = pp & 31;
        const int d0 = pr * 2;
        const int t  = t0 + m;
        const int b_ = t >> 11;
        const int s  = t & (SEQ - 1);
        float e = tile[m * 64 + d0]     + bias[g * 64 + d0];
        float o = tile[m * 64 + d0 + 1] + bias[g * 64 + d0 + 1];
        const size_t bh = (size_t)(b_ * NHEAD + h);
        if (which == 2) {
            vT[(bh * HDIM + d0) * SEQ + s]     = f2bf(e);
            vT[(bh * HDIM + d0 + 1) * SEQ + s] = f2bf(o);
        } else {
            float invf = __powf(10000.0f, -(float)d0 / (float)HDIM);
            float ang = (float)s * invf;
            float sn, cs;
            __sincosf(ang, &sn, &cs);
            float re = e * cs - o * sn;
            float ro = e * sn + o * cs;
            unsigned short* dst = (which == 0) ? q : k;
            size_t base = (bh * SEQ + s) * HDIM + d0;
            dst[base]     = f2bf(re);
            dst[base + 1] = f2bf(ro);
        }
    }
}

// ---------------------------------------------------------------------------
// Attention: per (b,h,16-query tile). Scores kept as bf16 in 64KB LDS,
// in-place softmax, then context GEMM against vT. One wave per block.
// grid = (128, 12, 2), block = 32.
// ---------------------------------------------------------------------------
__global__ __launch_bounds__(32)
void attn_kernel(const unsigned short* __restrict__ q,
                 const unsigned short* __restrict__ k,
                 const unsigned short* __restrict__ vT,
                 unsigned short* __restrict__ ctx) {
    __shared__ unsigned short sc[16 * SEQ];   // 64 KB

    const int qt = blockIdx.x;
    const int h  = blockIdx.y;
    const int b  = blockIdx.z;
    const int lane = threadIdx.x;
    const int Mloc = lane & 15, hi = lane >> 4;
    const size_t bh = (size_t)(b * NHEAD + h);

    // ---- scores = (Q K^T) / sqrt(D), stored bf16 in LDS ----
    // Q tile operands are loop-invariant: hoist them.
    const unsigned short* qrow = q + (bh * SEQ + qt * 16 + Mloc) * HDIM;
    const v16bf a0 = loadA(qrow, 0, hi);
    const v16bf a1 = loadA(qrow, 32, hi);

    for (int kt = 0; kt < SEQ / 16; ++kt) {
        const unsigned short* krow = k + (bh * SEQ + kt * 16 + Mloc) * HDIM;
        v8f acc = v8f{};
        acc = wmma_bf16(a0, loadB(krow, 0, hi), acc);
        acc = wmma_bf16(a1, loadB(krow, 32, hi), acc);
#pragma unroll
        for (int r = 0; r < 8; ++r)
            sc[(r + hi * 8) * SEQ + kt * 16 + Mloc] = f2bf(acc[r] * 0.125f);
    }
    __syncthreads();

    // ---- in-place softmax: 2 lanes per row, 1024 keys each ----
    {
        const int row = lane & 15;
        const int half = lane >> 4;
        unsigned short* prow = sc + row * SEQ + half * 1024;
        float mx = -1e30f;
        for (int j = 0; j < 1024; ++j) mx = fmaxf(mx, bf2f(prow[j]));
        mx = fmaxf(mx, __shfl_xor(mx, 16, 32));
        float sum = 0.0f;
        for (int j = 0; j < 1024; ++j) sum += __expf(bf2f(prow[j]) - mx);
        sum += __shfl_xor(sum, 16, 32);
        const float inv = 1.0f / sum;
        for (int j = 0; j < 1024; ++j)
            prow[j] = f2bf(__expf(bf2f(prow[j]) - mx) * inv);
    }
    __syncthreads();

    // ---- context = P @ V ----
    v8f acc[4] = {v8f{}, v8f{}, v8f{}, v8f{}};
    const unsigned short* prow = sc + (size_t)Mloc * SEQ;   // A rows (LDS)
    const unsigned short* vrow[4];
#pragma unroll
    for (int nt = 0; nt < 4; ++nt)
        vrow[nt] = vT + (bh * HDIM + nt * 16 + Mloc) * SEQ;

    for (int k0 = 0; k0 < SEQ; k0 += 32) {
        v16bf a = loadA(prow, k0, hi);
#pragma unroll
        for (int nt = 0; nt < 4; ++nt)
            acc[nt] = wmma_bf16(a, loadB(vrow[nt], k0, hi), acc[nt]);
    }
#pragma unroll
    for (int nt = 0; nt < 4; ++nt)
#pragma unroll
        for (int r = 0; r < 8; ++r) {
            const int s = qt * 16 + r + hi * 8;
            const int d = nt * 16 + Mloc;
            ctx[((size_t)(b * SEQ + s)) * HIDDEN + h * HDIM + d] = f2bf(acc[nt][r]);
        }
}

// ---------------------------------------------------------------------------
// Output projection: out[t,f] = sum_k ctx[t,k]*Wp[f,k] + pb[f], fp32 out.
// 64x64 tile per wave. grid = (64, 12), block = 32.
// ---------------------------------------------------------------------------
__global__ __launch_bounds__(32)
void proj_kernel(const unsigned short* __restrict__ ctxb,
                 const unsigned short* __restrict__ wb,
                 const float* __restrict__ bias,
                 float* __restrict__ out) {
    const int mt = blockIdx.x;
    const int g  = blockIdx.y;
    const int lane = threadIdx.x;
    const int Mloc = lane & 15, hi = lane >> 4;
    const int t0 = mt * 64;

    const unsigned short* arow[4];
#pragma unroll
    for (int i = 0; i < 4; ++i)
        arow[i] = ctxb + (size_t)(t0 + i * 16 + Mloc) * HIDDEN;
    const unsigned short* wrow[4];
#pragma unroll
    for (int nt = 0; nt < 4; ++nt)
        wrow[nt] = wb + (size_t)(g * 64 + nt * 16 + Mloc) * HIDDEN;

    v8f acc[4][4];
#pragma unroll
    for (int i = 0; i < 4; ++i)
#pragma unroll
        for (int nt = 0; nt < 4; ++nt) acc[i][nt] = v8f{};

    for (int k0 = 0; k0 < HIDDEN; k0 += 32) {
        v16bf a[4];
#pragma unroll
        for (int i = 0; i < 4; ++i) {
            a[i] = loadA(arow[i], k0, hi);
            __builtin_prefetch(arow[i] + k0 + 128, 0, 1);
        }
#pragma unroll
        for (int nt = 0; nt < 4; ++nt) {
            v16bf b = loadB(wrow[nt], k0, hi);
#pragma unroll
            for (int i = 0; i < 4; ++i)
                acc[i][nt] = wmma_bf16(a[i], b, acc[i][nt]);
        }
    }
#pragma unroll
    for (int i = 0; i < 4; ++i)
#pragma unroll
        for (int nt = 0; nt < 4; ++nt)
#pragma unroll
            for (int r = 0; r < 8; ++r) {
                const int t = t0 + i * 16 + r + hi * 8;
                const int f = g * 64 + nt * 16 + Mloc;
                out[(size_t)t * HIDDEN + f] = acc[i][nt][r] + bias[f];
            }
}

// ---------------------------------------------------------------------------
extern "C" void kernel_launch(void* const* d_in, const int* in_sizes, int n_in,
                              void* d_out, int out_size, void* d_ws, size_t ws_size,
                              hipStream_t stream) {
    const float* x      = (const float*)d_in[0];   // [2,2048,768]
    const float* qkv_w  = (const float*)d_in[1];   // [2304,768]
    const float* qkv_b  = (const float*)d_in[2];   // [2304]
    const float* proj_w = (const float*)d_in[3];   // [768,768]
    const float* proj_b = (const float*)d_in[4];   // [768]
    float* out = (float*)d_out;                    // [2,2048,768]

    const size_t NT = (size_t)BATCH * SEQ;         // 4096 tokens
    const size_t n_x  = NT * HIDDEN;
    const size_t n_qw = (size_t)3 * HIDDEN * HIDDEN;
    const size_t n_pw = (size_t)HIDDEN * HIDDEN;
    const size_t n_hd = (size_t)BATCH * NHEAD * SEQ * HDIM;

    unsigned char* ws = (unsigned char*)d_ws;
    size_t off = 0;
    unsigned short* xb  = (unsigned short*)(ws + off); off += n_x  * 2;
    unsigned short* qwb = (unsigned short*)(ws + off); off += n_qw * 2;
    unsigned short* pwb = (unsigned short*)(ws + off); off += n_pw * 2;
    unsigned short* qb  = (unsigned short*)(ws + off); off += n_hd * 2;
    unsigned short* kb  = (unsigned short*)(ws + off); off += n_hd * 2;
    unsigned short* vTb = (unsigned short*)(ws + off); off += n_hd * 2;
    unsigned short* ctx = (unsigned short*)(ws + off); off += n_x  * 2;
    (void)ws_size; (void)in_sizes; (void)n_in; (void)out_size;

    cvt_bf16_kernel<<<2048, 256, 0, stream>>>(x,      xb,  (int)n_x);
    cvt_bf16_kernel<<<2048, 256, 0, stream>>>(qkv_w,  qwb, (int)n_qw);
    cvt_bf16_kernel<<<1024, 256, 0, stream>>>(proj_w, pwb, (int)n_pw);

    qkv_rope_kernel<<<dim3(NT / 64, 36), 32, 0, stream>>>(xb, qwb, qkv_b,
                                                          qb, kb, vTb);

    attn_kernel<<<dim3(SEQ / 16, NHEAD, BATCH), 32, 0, stream>>>(qb, kb, vTb, ctx);

    proj_kernel<<<dim3(NT / 64, HIDDEN / 64), 32, 0, stream>>>(ctx, pwb, proj_b, out);
}
